// MambaBlock_69750268887305
// MI455X (gfx1250) — matrix operand order repaired and checked by reading, not compiled, not run on hardware
//
#include <hip/hip_runtime.h>
#include <hip/hip_bf16.h>
#include <cstdint>

// ---- problem constants (match reference) ----
#define DIMX    1024
#define D_STATE 16
#define D_CONVK 4
#define D_INNER 2048
#define DT_RANK 64
#define BSZ     2
#define SEQ     2048
#define NTOK    (BSZ * SEQ)              // 4096 tokens
#define XPROJ_N (DT_RANK + 2 * D_STATE)  // 96

#define USE_TDM 1

typedef __attribute__((ext_vector_type(16))) __bf16 v16bf;
typedef __attribute__((ext_vector_type(8)))  float  v8f;
typedef __attribute__((ext_vector_type(4)))  unsigned int u32x4;
typedef __attribute__((ext_vector_type(4)))  int    i32x4;
typedef __attribute__((ext_vector_type(8)))  int    i32x8;

__device__ __forceinline__ float sigmoidf_(float x) {
    return 1.f / (1.f + __expf(-x));
}

// Pack two 8-float runs (p0[0..7], p1[0..7]) into a bf16 WMMA fragment.
// A-frag: p1 = p0 + 8  (this code's order: K = kh*16+{0..15}, contiguous, same as B)
// B-frag: p1 = p0 + 8  (this code's order, same as A: K = kh*16+{0..15}, contiguous)
__device__ __forceinline__ v16bf pack_2x8(const float* p0, const float* p1) {
    float4 f0 = *(const float4*)(p0);
    float4 f1 = *(const float4*)(p0 + 4);
    float4 f2 = *(const float4*)(p1);
    float4 f3 = *(const float4*)(p1 + 4);
    v16bf r;
    r[0]  = (__bf16)f0.x; r[1]  = (__bf16)f0.y; r[2]  = (__bf16)f0.z; r[3]  = (__bf16)f0.w;
    r[4]  = (__bf16)f1.x; r[5]  = (__bf16)f1.y; r[6]  = (__bf16)f1.z; r[7]  = (__bf16)f1.w;
    r[8]  = (__bf16)f2.x; r[9]  = (__bf16)f2.y; r[10] = (__bf16)f2.z; r[11] = (__bf16)f2.w;
    r[12] = (__bf16)f3.x; r[13] = (__bf16)f3.y; r[14] = (__bf16)f3.z; r[15] = (__bf16)f3.w;
    return r;
}

// ===================== TDM: 2D panel load (rows x 32 f32) to LDS =====================
// Packs a Tensor DMA Descriptor (D#) per CDNA5 ISA 8.3/8.4:
//  - data_size = 4B, tile = 32 (K) x 128 (rows), tensor_dim0_stride = ld
//  - pad_enable: 4 DWORDs after every 32 DWORDs -> LDS row stride 36 floats
//    (makes ds_load_b128 fragment reads bank-conflict-free: 9r mod 16 bijective)
#if USE_TDM
__device__ __forceinline__ void tdm_load_panel(
    unsigned lds_byte_off, const float* gptr,
    unsigned rows_rem, unsigned cols_rem, unsigned ld_elems)
{
    unsigned long long ga = (unsigned long long)(uintptr_t)gptr;
    u32x4 g0;
    g0[0] = 1u;                                            // count=1 (valid D#)
    g0[1] = lds_byte_off;                                  // lds_addr
    g0[2] = (unsigned)(ga & 0xFFFFFFFFull);                // global_addr[31:0]
    g0[3] = (unsigned)((ga >> 32) & 0x01FFFFFFull)         // global_addr[56:32]
          | (2u << 30);                                    // type=2 ("image")
    i32x8 g1;
    g1[0] = (int)((2u << 16)                               // data_size = 4B
                | (1u << 20)                               // pad_enable
                | (4u << 22)                               // pad_interval: every 32 DW
                | (3u << 25));                             // pad_amount: 4 DW
    g1[1] = (int)((cols_rem & 0xFFFFu) << 16);             // tensor_dim0[15:0]
    g1[2] = (int)(((cols_rem >> 16) & 0xFFFFu)             // tensor_dim0[31:16]
                | ((rows_rem & 0xFFFFu) << 16));           // tensor_dim1[15:0]
    g1[3] = (int)(((rows_rem >> 16) & 0xFFFFu)             // tensor_dim1[31:16]
                | (32u << 16));                            // tile_dim0 = 32
    g1[4] = (int)128u;                                     // tile_dim1 = 128 (tile_dim2=0)
    g1[5] = (int)ld_elems;                                 // tensor_dim0_stride[31:0]
    g1[6] = 0;                                             // stride[47:32], dim1_stride lo
    g1[7] = 0;
    i32x4 z4 = {0, 0, 0, 0};
#if __clang_major__ >= 23
    i32x8 z8 = {0, 0, 0, 0, 0, 0, 0, 0};
    __builtin_amdgcn_tensor_load_to_lds(g0, g1, z4, z4, z8, 0);
#else
    __builtin_amdgcn_tensor_load_to_lds(g0, g1, z4, z4, 0);
#endif
}
#endif

// ===================== tiled GEMM: C[M,N] = A[M,K](lda) @ W[N,K](ldw)^T ==============
// 256 threads = 8 waves; block tile 128x128; K step 32; double-buffered LDS panels
// staged by the Tensor Data Mover; each wave computes 64x32 = 8 WMMA accumulators.
// LDS floats: As0 @0, As1 @4608, Bs0 @9216, Bs1 @13824 (row stride 36) = 73728 B.
#define LDSTRIDE 36
#define PANEL_F  4608
__global__ __launch_bounds__(256)
void gemm_tiled_wmma(const float* __restrict__ A, int lda,
                     const float* __restrict__ W, int ldw,
                     float* __restrict__ C, int ldc,
                     int M, int N, int K,
                     const float* __restrict__ bias, int epilogue)
{
    extern __shared__ float lds[];
    const int wave   = threadIdx.x >> 5;
    const int lane   = threadIdx.x & 31;
    const int lrow   = lane & 15;
    const int kh     = lane >> 4;
    const int wave_m = wave >> 2;                 // 0..1 -> 64-row half
    const int wave_n = wave & 3;                  // 0..3 -> 32-col quarter
    const int blockM = blockIdx.y * 128;
    const int blockN = blockIdx.x * 128;

    v8f acc[4][2] = {};
    const int nstages = K / 32;

#if USE_TDM
    if (wave == 0) {
        tdm_load_panel(0u,                 A + (size_t)blockM * lda,
                       (unsigned)(M - blockM), (unsigned)K, (unsigned)lda);
        tdm_load_panel((unsigned)(9216 * 4), W + (size_t)blockN * ldw,
                       (unsigned)(N - blockN), (unsigned)K, (unsigned)ldw);
    }
#endif

    for (int i = 0; i < nstages; ++i) {
        const int buf  = i & 1;
        const int nbuf = buf ^ 1;
#if USE_TDM
        if (wave == 0) {
            if (i + 1 < nstages) {
                const int k0n = (i + 1) * 32;
                tdm_load_panel((unsigned)(nbuf * PANEL_F * 4),
                               A + (size_t)blockM * lda + k0n,
                               (unsigned)(M - blockM), (unsigned)(K - k0n), (unsigned)lda);
                tdm_load_panel((unsigned)((9216 + nbuf * PANEL_F) * 4),
                               W + (size_t)blockN * ldw + k0n,
                               (unsigned)(N - blockN), (unsigned)(K - k0n), (unsigned)ldw);
                __builtin_amdgcn_s_wait_tensorcnt((short)2);   // current stage landed
            } else {
                __builtin_amdgcn_s_wait_tensorcnt((short)0);   // drain
            }
        }
        __syncthreads();
#else
        {   // fallback: per-thread staging, bank-conflict-free stride 36
            const int k0  = i * 32;
            const int row = threadIdx.x >> 1;
            const int hlf = threadIdx.x & 1;
            const float* ga = A + (size_t)(blockM + row) * lda + k0 + hlf * 16;
            const float* gb = W + (size_t)(blockN + row) * ldw + k0 + hlf * 16;
            float* sa = lds + buf * PANEL_F + row * LDSTRIDE + hlf * 16;
            float* sb = lds + 9216 + buf * PANEL_F + row * LDSTRIDE + hlf * 16;
#pragma unroll
            for (int j = 0; j < 4; ++j) {
                ((float4*)sa)[j] = ((const float4*)ga)[j];
                ((float4*)sb)[j] = ((const float4*)gb)[j];
            }
            __syncthreads();
        }
#endif
        const float* As = lds + buf * PANEL_F;
        const float* Bs = lds + 9216 + buf * PANEL_F;

        v16bf af[4];
#pragma unroll
        for (int mt = 0; mt < 4; ++mt) {
            const float* p = As + (wave_m * 64 + mt * 16 + lrow) * LDSTRIDE + kh * 16;
            af[mt] = pack_2x8(p, p + 8);
        }
        v16bf bfr[2];
#pragma unroll
        for (int nt = 0; nt < 2; ++nt) {
            const float* p = Bs + (wave_n * 32 + nt * 16 + lrow) * LDSTRIDE + kh * 16;
            bfr[nt] = pack_2x8(p, p + 8);
        }
#pragma unroll
        for (int mt = 0; mt < 4; ++mt)
#pragma unroll
            for (int nt = 0; nt < 2; ++nt)
                acc[mt][nt] = __builtin_amdgcn_wmma_f32_16x16x32_bf16(
                    false, af[mt], false, bfr[nt], (short)0, acc[mt][nt], false, false);
        __syncthreads();
    }

    // epilogue + store (C/D layout: lane{n=lane&15, kh}; VGPR r -> m = kh*8 + r)
#pragma unroll
    for (int mt = 0; mt < 4; ++mt) {
        const int mbase = blockM + wave_m * 64 + mt * 16 + kh * 8;
#pragma unroll
        for (int nt = 0; nt < 2; ++nt) {
            const int n  = blockN + wave_n * 32 + nt * 16 + lrow;
            const float bn = (epilogue == 1) ? bias[n] : 0.f;
#pragma unroll
            for (int r = 0; r < 8; ++r) {
                float v = acc[mt][nt][r];
                if (epilogue == 1) {
                    v += bn;
                    v = (v > 20.f) ? v : __logf(1.f + __expf(v));  // softplus
                }
                C[(size_t)(mbase + r) * ldc + n] = v;
            }
        }
    }
}

// ===================== simple per-wave GEMM (for N = 96 x-proj) ======================
__global__ __launch_bounds__(128)
void gemm_bf16_wmma(const float* __restrict__ A, int lda,
                    const float* __restrict__ W, int ldw,
                    float* __restrict__ C, int ldc,
                    int N, int K,
                    const float* __restrict__ bias, int epilogue)
{
    const int wave   = threadIdx.x >> 5;
    const int lane   = threadIdx.x & 31;
    const int n_tile = blockIdx.x * 4 + wave;
    const int m_tile = blockIdx.y;
    if (n_tile * 16 >= N) return;

    const int lrow  = lane & 15;
    const int khalf = lane >> 4;
    const float* aptr = A + (size_t)(m_tile * 16 + lrow) * lda + khalf * 16;
    const float* wptr = W + (size_t)(n_tile * 16 + lrow) * ldw + khalf * 16;

    v8f acc = {};
    for (int k0 = 0; k0 < K; k0 += 32) {
        if (k0 + 128 < K) {
            __builtin_prefetch(aptr + k0 + 128, 0, 1);
            __builtin_prefetch(wptr + k0 + 128, 0, 1);
        }
        v16bf av = pack_2x8(aptr + k0, aptr + k0 + 8);
        v16bf bv = pack_2x8(wptr + k0, wptr + k0 + 8);
        acc = __builtin_amdgcn_wmma_f32_16x16x32_bf16(
            false, av, false, bv, (short)0, acc, false, false);
    }
    const int   n     = n_tile * 16 + lrow;
    const int   mbase = m_tile * 16 + khalf * 8;
    const float bn    = (epilogue == 1) ? bias[n] : 0.f;
#pragma unroll
    for (int r = 0; r < 8; ++r) {
        float v = acc[r];
        if (epilogue == 1) {
            v += bn;
            v = (v > 20.f) ? v : __logf(1.f + __expf(v));
        }
        C[(size_t)(mbase + r) * ldc + n] = v;
    }
}

// ---- depthwise causal conv(k=4) + bias + SiLU : reads xc = xz[..., :D_INNER] ----
__global__ __launch_bounds__(256)
void conv_silu_kernel(const float* __restrict__ xz,
                      const float* __restrict__ cw,
                      const float* __restrict__ cb,
                      float* __restrict__ u)
{
    const size_t idx = (size_t)blockIdx.x * 256 + threadIdx.x;
    const int d = (int)(idx & (D_INNER - 1));
    const int l = (int)((idx >> 11) & (SEQ - 1));
    const int b = (int)(idx >> 22);
    float acc = cb[d];
#pragma unroll
    for (int j = 0; j < D_CONVK; ++j) {
        const int ll = l - (D_CONVK - 1) + j;
        if (ll >= 0)
            acc += cw[d * D_CONVK + j] *
                   xz[(size_t)(b * SEQ + ll) * (2 * D_INNER) + d];
    }
    u[idx] = acc * sigmoidf_(acc);
}

// ---- selective scan (sequential over L) + D skip + silu(z) gate, in place ----
__global__ __launch_bounds__(256)
void scan_gate_kernel(float* __restrict__ delta_y,
                      const float* __restrict__ u,
                      const float* __restrict__ xdbc,
                      const float* __restrict__ A_log,
                      const float* __restrict__ Dv,
                      const float* __restrict__ xz)
{
    __shared__ float sBC[2 * D_STATE];
    const int gid = blockIdx.x * 256 + threadIdx.x;
    const int b   = gid >> 11;
    const int d   = gid & (D_INNER - 1);

    float a[D_STATE], h[D_STATE];
#pragma unroll
    for (int s = 0; s < D_STATE; ++s) {
        a[s] = -__expf(A_log[d * D_STATE + s]);
        h[s] = 0.f;
    }
    const float dskip = Dv[d];

    for (int l = 0; l < SEQ; ++l) {
        const size_t row = (size_t)(b * SEQ + l);
        if (threadIdx.x < 2 * D_STATE)
            sBC[threadIdx.x] = xdbc[row * XPROJ_N + DT_RANK + threadIdx.x];
        __syncthreads();

        const size_t idx = row * D_INNER + d;
        const float dlt = delta_y[idx];
        const float ut  = u[idx];
        const float du  = dlt * ut;
        float y = 0.f;
#pragma unroll
        for (int s = 0; s < D_STATE; ++s) {
            const float dA = __expf(dlt * a[s]);
            h[s] = h[s] * dA + du * sBC[s];
            y += h[s] * sBC[D_STATE + s];
        }
        const float zz = xz[row * (2 * D_INNER) + D_INNER + d];
        delta_y[idx] = (y + ut * dskip) * (zz * sigmoidf_(zz));
        __syncthreads();
    }
}

// ---- LayerNorm over last dim (1024) ----
__global__ __launch_bounds__(256)
void layernorm_kernel(const float* __restrict__ in,
                      const float* __restrict__ gamma,
                      const float* __restrict__ beta,
                      float* __restrict__ out)
{
    __shared__ float ssum[256];
    __shared__ float ssq[256];
    const int row = blockIdx.x;
    const float* p = in + (size_t)row * DIMX;
    float s = 0.f, q = 0.f;
    for (int i = threadIdx.x; i < DIMX; i += 256) {
        const float v = p[i];
        s += v; q += v * v;
    }
    ssum[threadIdx.x] = s; ssq[threadIdx.x] = q;
    __syncthreads();
    for (int off = 128; off > 0; off >>= 1) {
        if (threadIdx.x < off) {
            ssum[threadIdx.x] += ssum[threadIdx.x + off];
            ssq[threadIdx.x]  += ssq[threadIdx.x + off];
        }
        __syncthreads();
    }
    const float mean = ssum[0] * (1.f / DIMX);
    const float var  = ssq[0] * (1.f / DIMX) - mean * mean;
    const float inv  = rsqrtf(var + 1e-5f);
    for (int i = threadIdx.x; i < DIMX; i += 256)
        out[(size_t)row * DIMX + i] = (p[i] - mean) * inv * gamma[i] + beta[i];
}

extern "C" void kernel_launch(void* const* d_in, const int* in_sizes, int n_in,
                              void* d_out, int out_size, void* d_ws, size_t ws_size,
                              hipStream_t stream)
{
    const float* x     = (const float*)d_in[0];
    const float* W_in  = (const float*)d_in[1];
    const float* convw = (const float*)d_in[2];
    const float* convb = (const float*)d_in[3];
    const float* W_xp  = (const float*)d_in[4];
    const float* W_dt  = (const float*)d_in[5];
    const float* b_dt  = (const float*)d_in[6];
    const float* A_log = (const float*)d_in[7];
    const float* Dv    = (const float*)d_in[8];
    const float* W_out = (const float*)d_in[9];
    const float* gamma = (const float*)d_in[10];
    const float* beta  = (const float*)d_in[11];
    float* out = (float*)d_out;

    // workspace (floats); ygate aliases delta, pre-LN output aliases xz
    float* ws    = (float*)d_ws;
    float* xz    = ws;                                    // NTOK x 4096
    float* u     = xz   + (size_t)NTOK * 2 * D_INNER;     // NTOK x 2048
    float* xdbc  = u    + (size_t)NTOK * D_INNER;         // NTOK x 96
    float* delta = xdbc + (size_t)NTOK * XPROJ_N;         // NTOK x 2048
    float* outp  = xz;

    const unsigned SHMEM = 73728;   // 4 x (128x36) f32 panels (double-buffered A & B)

    // 1) xz = x @ W_in^T   (TDM-staged tiled WMMA)
    gemm_tiled_wmma<<<dim3((2 * D_INNER) / 128, NTOK / 128), 256, SHMEM, stream>>>(
        x, DIMX, W_in, DIMX, xz, 2 * D_INNER, NTOK, 2 * D_INNER, DIMX, nullptr, 0);

    // 2) u = silu(depthwise_conv(xc) + conv_b)
    conv_silu_kernel<<<(unsigned)(((size_t)NTOK * D_INNER) / 256), 256, 0, stream>>>(
        xz, convw, convb, u);

    // 3) xdbc = u @ W_xproj^T   (N=96 -> per-wave kernel)
    gemm_bf16_wmma<<<dim3(2, NTOK / 16), 128, 0, stream>>>(
        u, D_INNER, W_xp, D_INNER, xdbc, XPROJ_N, XPROJ_N, D_INNER, nullptr, 0);

    // 4) delta = softplus(dt @ W_dt^T + b_dt)   (TDM-staged tiled WMMA, K=64)
    gemm_tiled_wmma<<<dim3(D_INNER / 128, NTOK / 128), 256, SHMEM, stream>>>(
        xdbc, XPROJ_N, W_dt, DT_RANK, delta, D_INNER, NTOK, D_INNER, DT_RANK, b_dt, 1);

    // 5) selective scan + D skip + silu(z) gate (in place on delta)
    scan_gate_kernel<<<(BSZ * D_INNER) / 256, 256, 0, stream>>>(
        delta, u, xdbc, A_log, Dv, xz);

    // 6) outp = ygate @ W_out^T   (TDM-staged tiled WMMA)
    gemm_tiled_wmma<<<dim3(DIMX / 128, NTOK / 128), 256, SHMEM, stream>>>(
        delta, D_INNER, W_out, D_INNER, outp, DIMX, NTOK, DIMX, D_INNER, nullptr, 0);

    // 7) LayerNorm -> d_out
    layernorm_kernel<<<NTOK, 256, 0, stream>>>(outp, gamma, beta, out);
}
